// CausalSelfAttention_65953517797905
// MI455X (gfx1250) — matrix operand order repaired and checked by reading, compiled, not run on hardware
//
#include <hip/hip_runtime.h>

// Causal self-attention forward for MI455X (gfx1250, wave32, WMMA bf16).
// B=4, S=2048, E=1024, H=16, D=64.
//
// Pipeline (all bf16 WMMA w/ f32 accum):
//   0) transpose+convert Wqkv->WqkvT[3E][E] bf16, Wproj->WprojT[E][E] bf16
//   1) qkv = x @ Wqkv + b : writes Q[B,H,S,D]*(D^-0.5), K[B,H,S,D], VT[B,H,D,S] (bf16)
//   2) flash attention computed transposed: St = K@Qt, Ot = VT@Pt
//      (C-layout of St feeds B-operand of the second WMMA with one shfl_xor(16))
//   3) out = O @ Wproj + b (f32 output)
//
// Round-2 change: software pipelining / double buffering so WMMAs consume
// fragments loaded one iteration earlier instead of stalling on fresh loads.

#define BB 4
#define SS 2048
#define EE 1024
#define HH 16
#define DD 64

typedef __attribute__((ext_vector_type(16))) __bf16 v16bf;
typedef __attribute__((ext_vector_type(8)))  float  v8f;

union Frag { v16bf v; uint4 q[2]; __bf16 e[16]; };
union F8   { float4 f[2]; float s[8]; };

static __device__ __forceinline__ float xor16(float x) {
  return __shfl_xor(x, 16, 32);
}

// ---------------------------------------------------------------------------
// 0) out[n*K + k] = (bf16) in[k*N + n]   (weight transpose + convert)
// ---------------------------------------------------------------------------
__global__ __launch_bounds__(256)
void k_transpose_cvt(const float* __restrict__ in, __bf16* __restrict__ out,
                     int K, int N) {
  size_t total = (size_t)K * N;
  for (size_t i = (size_t)blockIdx.x * blockDim.x + threadIdx.x; i < total;
       i += (size_t)gridDim.x * blockDim.x) {
    size_t n = i / (size_t)K;
    size_t k = i - n * (size_t)K;
    out[i] = (__bf16)in[k * (size_t)N + n];
  }
}

// ---------------------------------------------------------------------------
// 1) QKV GEMM: M = B*S = 8192, N = 3E = 3072, Kdim = E = 1024
//    grid(48, 64), block(256). Double-buffered over the k-loop.
// ---------------------------------------------------------------------------
__global__ __launch_bounds__(256)
void k_qkv(const float* __restrict__ x, const __bf16* __restrict__ WqkvT,
           const float* __restrict__ bqkv,
           __bf16* __restrict__ Qp, __bf16* __restrict__ Kp,
           __bf16* __restrict__ VTp) {
  const int lane = threadIdx.x & 31;
  const int col  = lane & 15;
  const int hl   = lane >> 4;   // half-lane select
  const int wave = threadIdx.x >> 5;
  const int mbase = (blockIdx.y * 8 + wave) * 16;
  const int nbase = blockIdx.x * 64;

  v8f acc[4] = {};
  const float* arow = x + (size_t)(mbase + col) * EE;

  F8   ar0[2], ar1[2];   // raw f32 A chunks, double buffered
  Frag bb[2][4];         // B fragments, double buffered

  auto load_stage = [&](int buf, int kb) {
    ar0[buf].f[0] = *(const float4*)(arow + kb + hl * 8);
    ar0[buf].f[1] = *(const float4*)(arow + kb + hl * 8 + 4);
    ar1[buf].f[0] = *(const float4*)(arow + kb + 16 + hl * 8);
    ar1[buf].f[1] = *(const float4*)(arow + kb + 16 + hl * 8 + 4);
#pragma unroll
    for (int t = 0; t < 4; ++t) {
      const __bf16* brow =
          WqkvT + (size_t)(nbase + t * 16 + col) * EE + kb + hl * 8;
      bb[buf][t].q[0] = *(const uint4*)(brow);
      bb[buf][t].q[1] = *(const uint4*)(brow + 16);
    }
  };

  load_stage(0, 0);
#pragma unroll 2
  for (int kb = 0; kb < EE; kb += 32) {
    const int cur = (kb >> 5) & 1;
    const int nxt = cur ^ 1;
    if (kb + 32 < EE) load_stage(nxt, kb + 32);   // prefetch next k-step

    Frag a;   // A fragment 16x32: row m=col, k = {hl*8+0..7, 16+hl*8+0..7}
#pragma unroll
    for (int j = 0; j < 8; ++j) {
      a.e[j]     = (__bf16)ar0[cur].s[j];
      a.e[8 + j] = (__bf16)ar1[cur].s[j];
    }
#pragma unroll
    for (int t = 0; t < 4; ++t)
      acc[t] = __builtin_amdgcn_wmma_f32_16x16x32_bf16(
          false, a.v, false, bb[cur][t].v, (short)0, acc[t], false, false);
  }

  // C layout: lane holds col n = col; rows m = v + 8*hl
#pragma unroll
  for (int t = 0; t < 4; ++t) {
    const int n    = nbase + t * 16 + col;
    const float bi = bqkv[n];
    const int sect = n >> 10;          // 0=Q 1=K 2=V
    const int nn   = n & 1023;
    const int h    = nn >> 6;
    const int d    = nn & 63;
#pragma unroll
    for (int v = 0; v < 8; ++v) {
      const int m = mbase + v + 8 * hl;
      const int b = m >> 11;
      const int s = m & 2047;
      const float val = acc[t][v] + bi;
      if (sect == 0)
        Qp[(((size_t)(b * HH + h)) * SS + s) * DD + d] = (__bf16)(val * 0.125f);
      else if (sect == 1)
        Kp[(((size_t)(b * HH + h)) * SS + s) * DD + d] = (__bf16)val;
      else
        VTp[(((size_t)(b * HH + h)) * DD + d) * SS + s] = (__bf16)val;
    }
  }
}

// ---------------------------------------------------------------------------
// 2) Flash attention (transposed). grid(16, B*H), block(256).
//    Each wave owns a 16-query tile, loops causal kv blocks of 32.
//    K fragments double-buffered across kv blocks; VT loads issued before
//    the softmax VALU work so exp/shuffles hide their latency.
// ---------------------------------------------------------------------------
__global__ __launch_bounds__(256)
void k_attn(const __bf16* __restrict__ Qp, const __bf16* __restrict__ Kp,
            const __bf16* __restrict__ VTp, __bf16* __restrict__ Op) {
  const int lane = threadIdx.x & 31;
  const int col  = lane & 15;
  const int hl   = lane >> 4;
  const int wave = threadIdx.x >> 5;
  const int bh   = blockIdx.y;                 // b*H + h
  const int tq   = blockIdx.x * 8 + wave;
  const int qbase = tq * 16;

  const size_t headQK = (size_t)bh * SS * DD;  // Q,K base for this head
  const size_t headVT = (size_t)bh * DD * SS;  // VT base

  // Q^T B-fragments (loaded once): bq[c] covers d = c*32..c*32+31,
  // per lane n = query col, k = {hl*8 + 0..7, 16 + hl*8 + 0..7} along d.
  Frag bq[2];
  {
    const __bf16* qrow = Qp + headQK + (size_t)(qbase + col) * DD;
#pragma unroll
    for (int c = 0; c < 2; ++c) {
      const __bf16* p = qrow + c * 32 + hl * 8;
      bq[c].q[0] = *(const uint4*)(p);
      bq[c].q[1] = *(const uint4*)(p + 16);
    }
  }

  v8f   oacc[4] = {};                 // Ot accumulators: 4 d-tiles x (16x16)
  float m_i = -1e30f, l_i = 0.0f;
  const int q_abs = qbase + col;      // this lane's query (C/B column)
  const int lastb = qbase & ~31;      // diagonal kv block

  Frag kf[2][2][2];                   // [buf][tile][chunk] K A-fragments
  auto load_k = [&](int buf, int kvb) {
#pragma unroll
    for (int t = 0; t < 2; ++t) {
      const __bf16* krow = Kp + headQK + (size_t)(kvb + t * 16 + col) * DD;
      kf[buf][t][0].q[0] = *(const uint4*)(krow + hl * 8);
      kf[buf][t][0].q[1] = *(const uint4*)(krow + 16 + hl * 8);
      kf[buf][t][1].q[0] = *(const uint4*)(krow + 32 + hl * 8);
      kf[buf][t][1].q[1] = *(const uint4*)(krow + 48 + hl * 8);
    }
  };

  load_k(0, 0);
#pragma unroll 2
  for (int kvb = 0; kvb <= lastb; kvb += 32) {
    const int cur = (kvb >> 5) & 1;
    const int nxt = cur ^ 1;
    if (kvb + 32 <= lastb) load_k(nxt, kvb + 32);   // prefetch next kv block

    // VT A-fragments for the CURRENT block: issue early, consumed after softmax
    Frag av[4];
#pragma unroll
    for (int t = 0; t < 4; ++t) {
      const __bf16* vrow = VTp + headVT + (size_t)(t * 16 + col) * SS;
      av[t].q[0] = *(const uint4*)(vrow + kvb + hl * 16);
      av[t].q[1] = *(const uint4*)(vrow + kvb + hl * 16 + 8);
    }

    // ---- St = K @ Qt : two 16x16 tiles (kv rows kvb+0..15, kvb+16..31)
    v8f st[2];
#pragma unroll
    for (int t = 0; t < 2; ++t) {
      v8f z = {};
      v8f p = __builtin_amdgcn_wmma_f32_16x16x32_bf16(
          false, kf[cur][t][0].v, false, bq[0].v, (short)0, z, false, false);
      st[t] = __builtin_amdgcn_wmma_f32_16x16x32_bf16(
          false, kf[cur][t][1].v, false, bq[1].v, (short)0, p, false, false);
    }

    // ---- causal mask on the diagonal block (lane rows: kv = kvb+t*16+8*hl+v)
    if (kvb == lastb) {
#pragma unroll
      for (int t = 0; t < 2; ++t)
#pragma unroll
        for (int v = 0; v < 8; ++v) {
          const int kv = kvb + t * 16 + 8 * hl + v;
          if (kv > q_abs) st[t][v] = -1e30f;
        }
    }

    // ---- online softmax (per-lane column stats; partner is lane^16)
    float mx = -1e30f;
#pragma unroll
    for (int v = 0; v < 8; ++v) mx = fmaxf(mx, fmaxf(st[0][v], st[1][v]));
    mx = fmaxf(mx, xor16(mx));
    const float m_new = fmaxf(m_i, mx);
    const float alpha = __expf(m_i - m_new);

    float p0[8], p1[8];
    float ls = 0.0f;
#pragma unroll
    for (int v = 0; v < 8; ++v) {
      p0[v] = __expf(st[0][v] - m_new);
      p1[v] = __expf(st[1][v] - m_new);
      ls += p0[v] + p1[v];
    }
    ls += xor16(ls);
    l_i = l_i * alpha + ls;
    m_i = m_new;
#pragma unroll
    for (int t = 0; t < 4; ++t) oacc[t] = oacc[t] * alpha;

    // ---- build Pt B-fragment: per lane n = q col, k = hl*16 + 0..15 (kv)
    // own C-layout gives kv {8*hl+v, 16+8*hl+v}; partner supplies the rest.
    Frag bp;
#pragma unroll
    for (int v = 0; v < 8; ++v) {
      const float other = xor16(hl ? p0[v] : p1[v]);
      const float klo   = hl ? other : p0[v];   // k = hl*16 + v
      const float khi   = hl ? p1[v] : other;   // k = hl*16 + 8 + v
      bp.e[v]     = (__bf16)klo;
      bp.e[8 + v] = (__bf16)khi;
    }

    // ---- Ot += VT @ Pt : 4 d-tiles, contraction k=32 over kv
#pragma unroll
    for (int t = 0; t < 4; ++t)
      oacc[t] = __builtin_amdgcn_wmma_f32_16x16x32_bf16(
          false, av[t].v, false, bp.v, (short)0, oacc[t], false, false);
  }

  // ---- store O[b][s][h*64+d] bf16; lane holds s = q_abs, d = t*16 + 8*hl + v
  const float inv_l = 1.0f / l_i;
  const int b = bh >> 4, h = bh & 15;
  __bf16* orow = Op + ((size_t)(b * SS + q_abs)) * EE + h * DD;
#pragma unroll
  for (int t = 0; t < 4; ++t)
#pragma unroll
    for (int v = 0; v < 8; ++v)
      orow[t * 16 + 8 * hl + v] = (__bf16)(oacc[t][v] * inv_l);
}

// ---------------------------------------------------------------------------
// 3) Output projection: out = O @ Wproj + bproj, f32 out. grid(16, 64).
//    Double-buffered over the k-loop.
// ---------------------------------------------------------------------------
__global__ __launch_bounds__(256)
void k_proj(const __bf16* __restrict__ Op, const __bf16* __restrict__ WprojT,
            const float* __restrict__ bproj, float* __restrict__ out) {
  const int lane = threadIdx.x & 31;
  const int col  = lane & 15;
  const int hl   = lane >> 4;
  const int wave = threadIdx.x >> 5;
  const int mbase = (blockIdx.y * 8 + wave) * 16;
  const int nbase = blockIdx.x * 64;

  v8f acc[4] = {};
  const __bf16* arow = Op + (size_t)(mbase + col) * EE;

  Frag aa[2];
  Frag bb[2][4];
  auto load_stage = [&](int buf, int kb) {
    aa[buf].q[0] = *(const uint4*)(arow + kb + hl * 8);
    aa[buf].q[1] = *(const uint4*)(arow + kb + 16 + hl * 8);
#pragma unroll
    for (int t = 0; t < 4; ++t) {
      const __bf16* brow =
          WprojT + (size_t)(nbase + t * 16 + col) * EE + kb + hl * 8;
      bb[buf][t].q[0] = *(const uint4*)(brow);
      bb[buf][t].q[1] = *(const uint4*)(brow + 16);
    }
  };

  load_stage(0, 0);
#pragma unroll 2
  for (int kb = 0; kb < EE; kb += 32) {
    const int cur = (kb >> 5) & 1;
    const int nxt = cur ^ 1;
    if (kb + 32 < EE) load_stage(nxt, kb + 32);
#pragma unroll
    for (int t = 0; t < 4; ++t)
      acc[t] = __builtin_amdgcn_wmma_f32_16x16x32_bf16(
          false, aa[cur].v, false, bb[cur][t].v, (short)0, acc[t], false, false);
  }

#pragma unroll
  for (int t = 0; t < 4; ++t) {
    const int n = nbase + t * 16 + col;
    const float bi = bproj[n];
#pragma unroll
    for (int v = 0; v < 8; ++v) {
      const int m = mbase + v + 8 * hl;
      out[(size_t)m * EE + n] = acc[t][v] + bi;
    }
  }
}

// ---------------------------------------------------------------------------
extern "C" void kernel_launch(void* const* d_in, const int* in_sizes, int n_in,
                              void* d_out, int out_size, void* d_ws,
                              size_t ws_size, hipStream_t stream) {
  (void)in_sizes; (void)n_in; (void)out_size; (void)ws_size;
  const float* x     = (const float*)d_in[0];
  const float* Wqkv  = (const float*)d_in[1];
  const float* bqkv  = (const float*)d_in[2];
  const float* Wproj = (const float*)d_in[3];
  const float* bproj = (const float*)d_in[4];
  float* out = (float*)d_out;

  char* ws = (char*)d_ws;
  __bf16* WqkvT  = (__bf16*)(ws);                  //  6,291,456 B
  __bf16* WprojT = (__bf16*)(ws + 6291456);        //  2,097,152 B
  __bf16* Qp     = (__bf16*)(ws + 8388608);        // 16,777,216 B
  __bf16* Kp     = (__bf16*)(ws + 25165824);       // 16,777,216 B
  __bf16* VTp    = (__bf16*)(ws + 41943040);       // 16,777,216 B
  __bf16* Op     = (__bf16*)(ws + 58720256);       // 16,777,216 B (total ~72 MB)

  k_transpose_cvt<<<1024, 256, 0, stream>>>(Wqkv, WqkvT, EE, 3 * EE);
  k_transpose_cvt<<<1024, 256, 0, stream>>>(Wproj, WprojT, EE, EE);

  k_qkv<<<dim3(48, 64), 256, 0, stream>>>(x, WqkvT, bqkv, Qp, Kp, VTp);
  k_attn<<<dim3(16, 64), 256, 0, stream>>>(Qp, Kp, VTp, Op);
  k_proj<<<dim3(16, 64), 256, 0, stream>>>(Op, WprojT, bproj, out);
}
